// GINConvGlobal_83545703841969
// MI455X (gfx1250) — matrix operand order repaired and checked by reading, not compiled, not run on hardware
//
#include <hip/hip_runtime.h>

// ---------------------------------------------------------------------------
// MegNet-style graph block for MI455X (gfx1250): bf16 WMMA GEMMs + fused
// gather / BN(batch-stats) / ReLU / scatter epilogues.
// Round 3: chunked ping-pong B-fragment buffers with sched_barrier fences so
// the backend cannot serialize load->wmma; 8 WMMAs issue per partial wait
// while the next chunk's 16 b128 loads remain in flight.
// ---------------------------------------------------------------------------

typedef __attribute__((ext_vector_type(16))) __bf16        v16bf;
typedef __attribute__((ext_vector_type(8)))  float         v8f;
typedef __attribute__((ext_vector_type(8)))  unsigned short us8;
typedef __attribute__((ext_vector_type(4)))  unsigned short us4;

#define HID   384   // MLP input width and hidden width
#define DD    128   // feature width D
#define TROWS 64    // rows per block tile
#define LSTR  392   // LDS row stride in ushorts (16B-aligned rows, bank-staggered)

__device__ __forceinline__ unsigned short f2bf(float f) {
  unsigned u = __float_as_uint(f);                       // round-to-nearest-even
  return (unsigned short)((u + 0x7FFFu + ((u >> 16) & 1u)) >> 16);
}
__device__ __forceinline__ float bf2f(unsigned short b) {
  return __uint_as_float(((unsigned)b) << 16);
}

union FragAB { v16bf v; us8 u[2]; };

#define WMMA_BF16(Acc, Af, Bf)                                               \
  Acc = __builtin_amdgcn_wmma_f32_16x16x32_bf16(false, (Af).v, false,        \
                                                (Bf).v, (short)0, Acc,       \
                                                false, false)

// K=384 sweep over two 16-wide column tiles (b0/b1 are N-major weight columns).
// 3 chunks x 4 K-steps, ping-pong B registers, sched_barrier(0) fences keep
// program order: [loads chunk ck+1][fence][8 wmma chunk ck][fence].
__device__ __forceinline__ void wmma_sweep(const unsigned short* __restrict__ b0,
                                           const unsigned short* __restrict__ b1,
                                           const FragAB* __restrict__ A,
                                           v8f& acc0, v8f& acc1) {
  FragAB Bb[2][8];
#pragma unroll
  for (int i = 0; i < 4; ++i) {
    int kb = i * 32;
    Bb[0][2 * i + 0].u[0] = *(const us8*)(b0 + kb);
    Bb[0][2 * i + 0].u[1] = *(const us8*)(b0 + kb + 8);
    Bb[0][2 * i + 1].u[0] = *(const us8*)(b1 + kb);
    Bb[0][2 * i + 1].u[1] = *(const us8*)(b1 + kb + 8);
  }
#pragma unroll
  for (int ck = 0; ck < 3; ++ck) {
    if (ck < 2) {
#pragma unroll
      for (int i = 0; i < 4; ++i) {
        int kb = ((ck + 1) * 4 + i) * 32;
        Bb[(ck + 1) & 1][2 * i + 0].u[0] = *(const us8*)(b0 + kb);
        Bb[(ck + 1) & 1][2 * i + 0].u[1] = *(const us8*)(b0 + kb + 8);
        Bb[(ck + 1) & 1][2 * i + 1].u[0] = *(const us8*)(b1 + kb);
        Bb[(ck + 1) & 1][2 * i + 1].u[1] = *(const us8*)(b1 + kb + 8);
      }
    }
    __builtin_amdgcn_sched_barrier(0);   // loads stay above, wmmas below
#pragma unroll
    for (int i = 0; i < 4; ++i) {
      WMMA_BF16(acc0, A[ck * 4 + i], Bb[ck & 1][2 * i + 0]);
      WMMA_BF16(acc1, A[ck * 4 + i], Bb[ck & 1][2 * i + 1]);
    }
    __builtin_amdgcn_sched_barrier(0);   // keep next chunk's loads after
  }
}

// ---------------------------------------------------------------------------
// Weight prep: Wt[n][k] = bf16(W[k][n])  (N-major so B-fragments are contiguous)
// ---------------------------------------------------------------------------
__global__ void k_transpose_w(const float* __restrict__ W,
                              unsigned short* __restrict__ Wt,
                              int K, int Ncols) {
  int idx = blockIdx.x * blockDim.x + threadIdx.x;
  if (idx >= K * Ncols) return;
  int n = idx / K, k = idx % K;
  Wt[(size_t)n * K + k] = f2bf(W[(size_t)k * Ncols + n]);
}

__global__ void k_copy(const float* __restrict__ a, float* __restrict__ b,
                       long long n) {
  long long i = (long long)blockIdx.x * blockDim.x + threadIdx.x;
  if (i < n) b[i] = a[i];
}

// sum_h[dst[i]][c] += h[src[i]][c]
__global__ void k_scatter_edge(const float* __restrict__ h,
                               const int* __restrict__ src,
                               const int* __restrict__ dst,
                               float* __restrict__ sum_h, long long total) {
  long long idx = (long long)blockIdx.x * blockDim.x + threadIdx.x;
  if (idx >= total) return;
  int i = (int)(idx >> 7), c = (int)(idx & (DD - 1));
  atomicAdd(&sum_h[(size_t)dst[i] * DD + c], h[(size_t)src[i] * DD + c]);
}

// seg[gid[i]][c] += 0.5f * sum_e[i][c]
__global__ void k_scatter_graph(const float* __restrict__ sum_e,
                                const int* __restrict__ gid,
                                float* __restrict__ seg, long long total) {
  long long idx = (long long)blockIdx.x * blockDim.x + threadIdx.x;
  if (idx >= total) return;
  int i = (int)(idx >> 7), c = (int)(idx & (DD - 1));
  atomicAdd(&seg[(size_t)gid[i] * DD + c], 0.5f * sum_e[(size_t)i * DD + c]);
}

// ---------------------------------------------------------------------------
// GEMM1: gather 384-wide input row -> LDS (bf16), z = X @ W1 (bias cancels in BN)
// mode 0: bond  X = [h[src]+h[dst], e, u[gid[src]]]
// mode 1: atom  X = [sum_h, sum_e, u[gid]]
// mode 2: glob  X = [sum_h_g, sum_e_g(0.5 pre-applied), u]
// ---------------------------------------------------------------------------
__global__ __launch_bounds__(128) void k_gather_gemm1(
    int mode, int R,
    const float* __restrict__ h, const float* __restrict__ e,
    const float* __restrict__ u,
    const int* __restrict__ src, const int* __restrict__ dst,
    const int* __restrict__ gid,
    const float* __restrict__ sum_h, const float* __restrict__ sum_e,
    const float* __restrict__ shg, const float* __restrict__ seg,
    const unsigned short* __restrict__ Wt, unsigned short* __restrict__ z) {
  __shared__ unsigned short Xs[TROWS * LSTR];
  const int tid = threadIdx.x;
  const int rowBase = blockIdx.x * TROWS;

  // ---- stage: gather + fp32->bf16 into LDS (2 threads per row, 192 cols each)
  {
    const int r = tid >> 1;
    const int half = tid & 1;
    const int grow = rowBase + r;
    const float* pa = h; const float* pb = h;
    const float* pe = e; const float* pu = u;
    if (grow < R) {
      if (mode == 0) {
        int s = src[grow], d = dst[grow];
        pa = h + (size_t)s * DD;
        pb = h + (size_t)d * DD;
        pe = e + (size_t)grow * DD;
        pu = u + (size_t)gid[s] * DD;
      } else if (mode == 1) {
        pa = sum_h + (size_t)grow * DD;
        pe = sum_e + (size_t)grow * DD;
        pu = u + (size_t)gid[grow] * DD;
      } else {
        pa = shg + (size_t)grow * DD;
        pe = seg + (size_t)grow * DD;
        pu = u + (size_t)grow * DD;
      }
    }
    for (int j = 0; j < 48; ++j) {
      int c0 = half * 192 + j * 4;
      int sg = c0 >> 7, c = c0 & (DD - 1);
      float x0 = 0.f, x1 = 0.f, x2 = 0.f, x3 = 0.f;
      if (grow < R) {
        const float* p = (sg == 0) ? pa : ((sg == 1) ? pe : pu);
        float4 a = *(const float4*)(p + c);
        x0 = a.x; x1 = a.y; x2 = a.z; x3 = a.w;
        if (sg == 0 && mode == 0) {
          float4 b = *(const float4*)(pb + c);
          x0 += b.x; x1 += b.y; x2 += b.z; x3 += b.w;
        }
      }
      us4 o; o.x = f2bf(x0); o.y = f2bf(x1); o.z = f2bf(x2); o.w = f2bf(x3);
      *(us4*)(&Xs[r * LSTR + c0]) = o;
    }
  }
  __syncthreads();

  // ---- WMMA: wave owns 16 rows; 24 col tiles processed 2 at a time ----
  const int lane = tid & 31;
  const int wave = tid >> 5;
  const int lr = lane & 15, lh = lane >> 4;
  const int wrow = wave * 16;

  FragAB A[12];
  const unsigned short* arow = &Xs[(wrow + lr) * LSTR];
#pragma unroll
  for (int kt = 0; kt < 12; ++kt) {
    int kb = kt * 32 + lh * 16;
    A[kt].u[0] = *(const us8*)(arow + kb);        // K = kb .. kb+7
    A[kt].u[1] = *(const us8*)(arow + kb + 8);    // K = kb+8 .. kb+15
  }
  for (int nt = 0; nt < HID / 16; nt += 2) {
    const unsigned short* b0 = Wt + (size_t)(nt * 16 + lr) * HID + lh * 16;
    const unsigned short* b1 = b0 + (size_t)16 * HID;
    v8f acc0 = {}, acc1 = {};
    wmma_sweep(b0, b1, A, acc0, acc1);
    const int rb = rowBase + wrow + lh * 8;
    const int col0 = nt * 16 + lr;
#pragma unroll
    for (int m = 0; m < 8; ++m) {
      int rr = rb + m;
      if (rr < R) {
        z[(size_t)rr * HID + col0]      = f2bf(acc0[m]);
        z[(size_t)rr * HID + col0 + 16] = f2bf(acc1[m]);
      }
    }
  }
}

// ---------------------------------------------------------------------------
// Column statistics over z (for batch-mode BN): sum and sum-of-squares.
// grid = (ceil(R/1024), 3), block = 128; thread owns one column.
// ---------------------------------------------------------------------------
__global__ void k_colstats(const unsigned short* __restrict__ z, int R,
                           float* __restrict__ s, float* __restrict__ ss) {
  int col = blockIdx.y * 128 + threadIdx.x;
  int r0 = blockIdx.x * 1024;
  int r1 = r0 + 1024; if (r1 > R) r1 = R;
  float a = 0.f, b = 0.f;
  for (int r = r0; r < r1; ++r) {
    float v = bf2f(z[(size_t)r * HID + col]);
    a += v; b += v * v;
  }
  atomicAdd(&s[col], a);
  atomicAdd(&ss[col], b);
}

__global__ void k_bnfinal(const float* __restrict__ s, const float* __restrict__ ss,
                          const float* __restrict__ g1, const float* __restrict__ beta1,
                          float* __restrict__ scale, float* __restrict__ shift,
                          float invR) {
  int c = blockIdx.x * blockDim.x + threadIdx.x;
  if (c >= HID) return;
  float mu = s[c] * invR;
  float var = ss[c] * invR - mu * mu;
  float sc = g1[c] * rsqrtf(var + 1e-5f);
  scale[c] = sc;
  shift[c] = beta1[c] - mu * sc;
}

// ---------------------------------------------------------------------------
// GEMM2: y = relu(z*scale+shift) staged to LDS as bf16; out = y @ W2 + b2.
// Optional fused scatter: scat[sidx[row]] += out[row]  (segment sums).
// ---------------------------------------------------------------------------
__global__ __launch_bounds__(128) void k_norm_gemm2(
    int R, const unsigned short* __restrict__ z,
    const float* __restrict__ scale, const float* __restrict__ shift,
    const unsigned short* __restrict__ Wt, const float* __restrict__ b2,
    float* __restrict__ out, const int* __restrict__ sidx,
    float* __restrict__ scat) {
  __shared__ unsigned short Ys[TROWS * LSTR];
  const int tid = threadIdx.x;
  const int rowBase = blockIdx.x * TROWS;
  {
    const int r = tid >> 1;
    const int half = tid & 1;
    const int grow = rowBase + r;  // z padded to a TROWS multiple: always in-bounds
    const unsigned short* zp = z + (size_t)grow * HID;
    for (int j = 0; j < 48; ++j) {
      int c0 = half * 192 + j * 4;
      us4 zi = *(const us4*)(zp + c0);
      float4 sc = *(const float4*)(scale + c0);
      float4 sf = *(const float4*)(shift + c0);
      us4 o;
      o.x = f2bf(fmaxf(bf2f(zi.x) * sc.x + sf.x, 0.f));
      o.y = f2bf(fmaxf(bf2f(zi.y) * sc.y + sf.y, 0.f));
      o.z = f2bf(fmaxf(bf2f(zi.z) * sc.z + sf.z, 0.f));
      o.w = f2bf(fmaxf(bf2f(zi.w) * sc.w + sf.w, 0.f));
      *(us4*)(&Ys[r * LSTR + c0]) = o;
    }
  }
  __syncthreads();

  const int lane = tid & 31;
  const int wave = tid >> 5;
  const int lr = lane & 15, lh = lane >> 4;
  const int wrow = wave * 16;

  FragAB A[12];
  const unsigned short* arow = &Ys[(wrow + lr) * LSTR];
#pragma unroll
  for (int kt = 0; kt < 12; ++kt) {
    int kb = kt * 32 + lh * 16;
    A[kt].u[0] = *(const us8*)(arow + kb);
    A[kt].u[1] = *(const us8*)(arow + kb + 8);
  }
  for (int nt = 0; nt < DD / 16; nt += 2) {
    const unsigned short* b0 = Wt + (size_t)(nt * 16 + lr) * HID + lh * 16;
    const unsigned short* b1 = b0 + (size_t)16 * HID;
    v8f acc0 = {}, acc1 = {};
    wmma_sweep(b0, b1, A, acc0, acc1);
    const int col0 = nt * 16 + lr;
    const float bias0 = b2[col0];
    const float bias1 = b2[col0 + 16];
    const int rb = rowBase + wrow + lh * 8;
#pragma unroll
    for (int m = 0; m < 8; ++m) {
      int rr = rb + m;
      if (rr < R) {
        float v0 = acc0[m] + bias0;
        float v1 = acc1[m] + bias1;
        out[(size_t)rr * DD + col0]      = v0;
        out[(size_t)rr * DD + col0 + 16] = v1;
        if (scat) {
          size_t sb = (size_t)sidx[rr] * DD;
          atomicAdd(&scat[sb + col0],      v0);
          atomicAdd(&scat[sb + col0 + 16], v1);
        }
      }
    }
  }
}

// ---------------------------------------------------------------------------
// Host orchestration
// ---------------------------------------------------------------------------
extern "C" void kernel_launch(void* const* d_in, const int* in_sizes, int n_in,
                              void* d_out, int out_size, void* d_ws, size_t ws_size,
                              hipStream_t stream) {
  const float* h  = (const float*)d_in[0];
  const float* e  = (const float*)d_in[1];
  const float* u  = (const float*)d_in[2];
  const int* src  = (const int*)d_in[3];
  const int* dst  = (const int*)d_in[4];
  const int* gid  = (const int*)d_in[5];
  const float* W1[3]  = { (const float*)d_in[6],  (const float*)d_in[12], (const float*)d_in[18] };
  const float* g1[3]  = { (const float*)d_in[8],  (const float*)d_in[14], (const float*)d_in[20] };
  const float* be1[3] = { (const float*)d_in[9],  (const float*)d_in[15], (const float*)d_in[21] };
  const float* W2[3]  = { (const float*)d_in[10], (const float*)d_in[16], (const float*)d_in[22] };
  const float* b2[3]  = { (const float*)d_in[11], (const float*)d_in[17], (const float*)d_in[23] };

  const int N = in_sizes[0] / DD;
  const int E = in_sizes[1] / DD;
  const int G = in_sizes[2] / DD;

  // ---- workspace carve-out (256B aligned) ----
  char* p = (char*)d_ws;
  auto carve = [&](size_t bytes) -> void* {
    void* q = (void*)p;
    p += (bytes + 255) & ~(size_t)255;
    return q;
  };
  unsigned short* wt1[3]; unsigned short* wt2[3];
  float *cs[3], *css[3], *sca[3], *shi[3];
  for (int i = 0; i < 3; ++i) wt1[i] = (unsigned short*)carve((size_t)HID * HID * 2);
  for (int i = 0; i < 3; ++i) wt2[i] = (unsigned short*)carve((size_t)DD * HID * 2);
  for (int i = 0; i < 3; ++i) { cs[i]  = (float*)carve(HID * 4); css[i] = (float*)carve(HID * 4); }
  for (int i = 0; i < 3; ++i) { sca[i] = (float*)carve(HID * 4); shi[i] = (float*)carve(HID * 4); }
  float* sum_h = (float*)carve((size_t)N * DD * 4);
  float* sum_e = (float*)carve((size_t)N * DD * 4);
  float* shg   = (float*)carve((size_t)G * DD * 4);
  float* seg   = (float*)carve((size_t)G * DD * 4);
  size_t padE  = ((size_t)E + TROWS - 1) & ~(size_t)(TROWS - 1);
  unsigned short* z = (unsigned short*)carve(padE * HID * 2);

  // ---- zero accumulators ----
  hipMemsetAsync(sum_e, 0, (size_t)N * DD * 4, stream);
  hipMemsetAsync(shg,   0, (size_t)G * DD * 4, stream);
  hipMemsetAsync(seg,   0, (size_t)G * DD * 4, stream);
  for (int i = 0; i < 3; ++i) {
    hipMemsetAsync(cs[i],  0, HID * 4, stream);
    hipMemsetAsync(css[i], 0, HID * 4, stream);
  }

  // ---- weight prep (bf16, N-major) ----
  for (int i = 0; i < 3; ++i) {
    k_transpose_w<<<(HID * HID + 255) / 256, 256, 0, stream>>>(W1[i], wt1[i], HID, HID);
    k_transpose_w<<<(HID * DD + 255) / 256, 256, 0, stream>>>(W2[i], wt2[i], HID, DD);
  }

  // ---- sum_h = h + segment_sum(h[src] by dst) ----
  long long nh = (long long)N * DD;
  k_copy<<<(unsigned)((nh + 255) / 256), 256, 0, stream>>>(h, sum_h, nh);
  long long ne = (long long)E * DD;
  k_scatter_edge<<<(unsigned)((ne + 255) / 256), 256, 0, stream>>>(h, src, dst, sum_h, ne);

  float* outh = (float*)d_out;
  float* oute = outh + (size_t)N * DD;
  float* outu = oute + (size_t)E * DD;

  // ---- bond MLP (rows = E); fused scatter e_new -> sum_e[dst] ----
  int nb = (E + TROWS - 1) / TROWS;
  k_gather_gemm1<<<nb, 128, 0, stream>>>(0, E, h, e, u, src, dst, gid,
                                         sum_h, sum_e, shg, seg, wt1[0], z);
  k_colstats<<<dim3((E + 1023) / 1024, 3), 128, 0, stream>>>(z, E, cs[0], css[0]);
  k_bnfinal<<<3, 128, 0, stream>>>(cs[0], css[0], g1[0], be1[0], sca[0], shi[0], 1.0f / E);
  k_norm_gemm2<<<nb, 128, 0, stream>>>(E, z, sca[0], shi[0], wt2[0], b2[0],
                                       oute, dst, sum_e);

  // ---- atom MLP (rows = N); fused scatter h_new -> sum_h_g[graph_id] ----
  int na = (N + TROWS - 1) / TROWS;
  k_gather_gemm1<<<na, 128, 0, stream>>>(1, N, h, e, u, src, dst, gid,
                                         sum_h, sum_e, shg, seg, wt1[1], z);
  k_colstats<<<dim3((N + 1023) / 1024, 3), 128, 0, stream>>>(z, N, cs[1], css[1]);
  k_bnfinal<<<3, 128, 0, stream>>>(cs[1], css[1], g1[1], be1[1], sca[1], shi[1], 1.0f / N);
  k_norm_gemm2<<<na, 128, 0, stream>>>(N, z, sca[1], shi[1], wt2[1], b2[1],
                                       outh, gid, shg);

  // ---- sum_e_g = 0.5 * segment_sum(sum_e by graph_id) ----
  k_scatter_graph<<<(unsigned)((nh + 255) / 256), 256, 0, stream>>>(sum_e, gid, seg, nh);

  // ---- global MLP (rows = G) ----
  int ng = (G + TROWS - 1) / TROWS;
  k_gather_gemm1<<<ng, 128, 0, stream>>>(2, G, h, e, u, src, dst, gid,
                                         sum_h, sum_e, shg, seg, wt1[2], z);
  k_colstats<<<dim3((G + 1023) / 1024, 3), 128, 0, stream>>>(z, G, cs[2], css[2]);
  k_bnfinal<<<3, 128, 0, stream>>>(cs[2], css[2], g1[2], be1[2], sca[2], shi[2], 1.0f / G);
  k_norm_gemm2<<<ng, 128, 0, stream>>>(G, z, sca[2], shi[2], wt2[2], b2[2],
                                       outu, nullptr, nullptr);
}
